// CrossAttention_25786983645652
// MI455X (gfx1250) — matrix operand order repaired and checked by reading, not compiled, not run on hardware
//
#include <hip/hip_runtime.h>
#include <hip/hip_bf16.h>

#define BB      8
#define NTOK    4096
#define DD      256
#define DD2     512
#define NHEADS  8
#define DKH     32
#define EPSV    1e-5f

typedef __attribute__((ext_vector_type(16))) __bf16 v16bf;
typedef __attribute__((ext_vector_type(8)))  float  v8f;
typedef __attribute__((ext_vector_type(4)))  unsigned v4u;
typedef __attribute__((ext_vector_type(8)))  unsigned v8u;

static __device__ __forceinline__ unsigned short f2bfu(float f) {
    union { float f; unsigned u; } c; c.f = f;
    unsigned u = c.u;
    unsigned r = u + 0x7FFFu + ((u >> 16) & 1u);   // round-to-nearest-even
    return (unsigned short)(r >> 16);
}

// ---- WMMA fragment-order swizzles (inverse of ISA 7.12.2 VGPR layouts) ----
// A-tile 16x32 bf16: element (row r, k) -> slot = lane*16 + e
static __device__ __forceinline__ int aSwz(int r, int kk) {
    int half = (kk >> 3) & 1;
    int e = (kk & 7) | ((kk & 16) >> 1);
    return ((half << 4) + r) * 16 + e;
}
// B-tile 32x16 bf16: element (k, col n) -> slot = lane*16 + e
static __device__ __forceinline__ int bSwz(int kk, int n) {
    return ((((kk >> 3) & 1) << 4) + n) * 16 + ((kk & 7) | ((kk & 16) >> 1));
}
// One contiguous 32B LDS read per lane = the whole v16bf fragment
static __device__ __forceinline__ v16bf ldFrag(const unsigned short* base, int lane) {
    return *reinterpret_cast<const v16bf*>(base + (lane << 4));
}

// ---- gfx1250 async global->LDS 16B copy (ASYNCcnt-tracked) ----
static __device__ __forceinline__ void async16(unsigned lds, unsigned long long sbase, int voff) {
    asm volatile("global_load_async_to_lds_b128 %0, %1, %2"
                 :: "v"(lds), "v"(voff), "s"(sbase) : "memory");
}
static __device__ __forceinline__ void wait_async0() {
    asm volatile("s_wait_asynccnt 0x0" ::: "memory");
}

// ---- gfx1250 Tensor Data Mover: 1D 32KB tile, global -> LDS (TENSORcnt) ----
// D# per ISA 8.3/8.4: count=1, data_size=8B, tensor_dim0=tile_dim0=4096 units
static __device__ __forceinline__ void tdm_load_32k(unsigned lds, unsigned long long gaddr) {
    v4u g0;
    g0[0] = 1u;                                        // count=1, user mode, no gather
    g0[1] = lds;                                       // lds_addr (bytes)
    g0[2] = (unsigned)gaddr;                           // global_addr[31:0]
    g0[3] = (unsigned)(gaddr >> 32) | (2u << 30);      // global_addr[56:32] | type=2
    v8u g1;
    g1[0] = 3u << 16;                                  // wg_mask=0, data_size=3 (8B)
    g1[1] = 0x1000u << 16;                             // tensor_dim0 = 4096 (lo16 @ bits63:48)
    g1[2] = 1u << 16;                                  // tensor_dim0 hi=0, tensor_dim1 = 1
    g1[3] = 0x1000u << 16;                             // tile_dim0 = 4096
    g1[4] = 0u;                                        // tile_dim1 = tile_dim2 = 0 (unused)
    g1[5] = 4096u;                                     // tensor_dim0_stride
    g1[6] = 0u;
    g1[7] = 0u;
    asm volatile("tensor_load_to_lds %0, %1, null, null"
                 :: "s"(g0), "s"(g1) : "memory");
    asm volatile("s_wait_tensorcnt 0x0" ::: "memory");
}

// ---------------------------------------------------------------------------
// K1: n1 = LN(x1 @ W_lin + b_lin) -> bf16, stored PRE-SWIZZLED in WMMA A-order
//     n1s[rowTile(2048)][kc(8)][512]
// block = 128 thr (4 waves): 16 rows x 256 cols, K=512
// ---------------------------------------------------------------------------
__global__ __launch_bounds__(128) void k_gemm1_ln(
    const float* __restrict__ x1, const float* __restrict__ w,
    const float* __restrict__ bias, const float* __restrict__ g,
    const float* __restrict__ bb, unsigned short* __restrict__ n1s)
{
    __shared__ __align__(32) unsigned short aF[512];
    __shared__ __align__(32) unsigned short bF[16 * 512];
    __shared__ float rowsum[16], rowsq[16], rmean[16], rrstd[16];

    const int tid = threadIdx.x;
    const int wv = tid >> 5, lane = tid & 31;
    const int half = lane >> 4, l15 = lane & 15;
    const int m0 = blockIdx.x * 16;

    v8f zero = {0.f,0.f,0.f,0.f,0.f,0.f,0.f,0.f};
    v8f acc[4];
    #pragma unroll
    for (int t = 0; t < 4; ++t) acc[t] = zero;

    for (int k0 = 0; k0 < DD2; k0 += 32) {
        for (int i = tid; i < 16 * 32; i += 128) {
            int r = i >> 5, kk = i & 31;
            aF[aSwz(r, kk)] = f2bfu(x1[(m0 + r) * DD2 + k0 + kk]);
        }
        for (int i = tid; i < 32 * DD; i += 128) {
            int kk = i >> 8, o = i & (DD - 1);
            bF[(o >> 4) * 512 + bSwz(kk, o & 15)] = f2bfu(w[(k0 + kk) * DD + o]);
        }
        __syncthreads();
        v16bf af = ldFrag(aF, lane);
        #pragma unroll
        for (int t = 0; t < 4; ++t) {
            v16bf bf = ldFrag(&bF[(wv * 4 + t) * 512], lane);
            acc[t] = __builtin_amdgcn_wmma_f32_16x16x32_bf16(
                false, af, false, bf, (short)0, acc[t], false, false);
        }
        __syncthreads();
    }

    if (tid < 16) { rowsum[tid] = 0.f; rowsq[tid] = 0.f; }
    __syncthreads();

    float vals[4][8];
    #pragma unroll
    for (int t = 0; t < 4; ++t) {
        int o = wv * 64 + t * 16 + l15;
        #pragma unroll
        for (int r = 0; r < 8; ++r) vals[t][r] = acc[t][r] + bias[o];
    }
    #pragma unroll
    for (int r = 0; r < 8; ++r) {
        int m = r + half * 8;
        float s = 0.f, s2 = 0.f;
        #pragma unroll
        for (int t = 0; t < 4; ++t) { float v = vals[t][r]; s += v; s2 += v * v; }
        atomicAdd(&rowsum[m], s);
        atomicAdd(&rowsq[m], s2);
    }
    __syncthreads();
    if (tid < 16) {
        float mu = rowsum[tid] / (float)DD;
        float var = rowsq[tid] / (float)DD - mu * mu;
        rmean[tid] = mu; rrstd[tid] = rsqrtf(var + EPSV);
    }
    __syncthreads();
    // write n1 pre-swizzled for K6's A tiles: rowTile = blockIdx.x, kc = o>>5
    #pragma unroll
    for (int t = 0; t < 4; ++t) {
        int o = wv * 64 + t * 16 + l15;
        #pragma unroll
        for (int r = 0; r < 8; ++r) {
            int m = r + half * 8;
            float v = (vals[t][r] - rmean[m]) * rrstd[m] * g[o] + bb[o];
            long idx = ((long)blockIdx.x * 8 + (o >> 5)) * 512 + aSwz(m, o & 31);
            n1s[idx] = f2bfu(v);
        }
    }
}

// ---------------------------------------------------------------------------
// K2: n2 = LN(x2), write transposed kq_t[b, c, n]  (wave per token)
// ---------------------------------------------------------------------------
__global__ __launch_bounds__(256) void k_ln2_t(
    const float* __restrict__ x2, const float* __restrict__ g,
    const float* __restrict__ bb, float* __restrict__ kq_t)
{
    int tid = threadIdx.x, wv = tid >> 5, lane = tid & 31;
    long tok = (long)blockIdx.x * 8 + wv;
    int b = (int)(tok >> 12), n = (int)(tok & 4095);
    float v[8]; float s = 0.f, s2 = 0.f;
    #pragma unroll
    for (int j = 0; j < 8; ++j) {
        v[j] = x2[tok * DD + lane + j * 32];
        s += v[j]; s2 += v[j] * v[j];
    }
    #pragma unroll
    for (int off = 16; off; off >>= 1) {
        s  += __shfl_xor(s,  off, 32);
        s2 += __shfl_xor(s2, off, 32);
    }
    float mu = s / (float)DD;
    float rstd = rsqrtf(s2 / (float)DD - mu * mu + EPSV);
    #pragma unroll
    for (int j = 0; j < 8; ++j) {
        int c = lane + j * 32;
        kq_t[((long)(b * DD + c)) * NTOK + n] = (v[j] - mu) * rstd * g[c] + bb[c];
    }
}

// ---------------------------------------------------------------------------
// K3: k = softmax over tokens (row of 4096), block per (b,c) row
// ---------------------------------------------------------------------------
__global__ __launch_bounds__(256) void k_softmax_tok(
    const float* __restrict__ kq_t, float* __restrict__ k_t)
{
    __shared__ float redm[8], reds[8];
    int tid = threadIdx.x, lane = tid & 31, wv = tid >> 5;
    long row = blockIdx.x;
    const float* p = kq_t + row * NTOK;
    float v[16]; float mx = -1e30f;
    #pragma unroll
    for (int j = 0; j < 16; ++j) { v[j] = p[tid + j * 256]; mx = fmaxf(mx, v[j]); }
    #pragma unroll
    for (int off = 16; off; off >>= 1) mx = fmaxf(mx, __shfl_xor(mx, off, 32));
    if (lane == 0) redm[wv] = mx;
    __syncthreads();
    if (tid == 0) { float m = redm[0]; for (int i = 1; i < 8; ++i) m = fmaxf(m, redm[i]); redm[0] = m; }
    __syncthreads();
    mx = redm[0];
    float s = 0.f;
    #pragma unroll
    for (int j = 0; j < 16; ++j) { v[j] = __expf(v[j] - mx); s += v[j]; }
    #pragma unroll
    for (int off = 16; off; off >>= 1) s += __shfl_xor(s, off, 32);
    if (lane == 0) reds[wv] = s;
    __syncthreads();
    if (tid == 0) { float m = 0.f; for (int i = 0; i < 8; ++i) m += reds[i]; reds[0] = m; }
    __syncthreads();
    float inv = 1.f / reds[0];
    float* po = k_t + row * NTOK;
    #pragma unroll
    for (int j = 0; j < 16; ++j) po[tid + j * 256] = v[j] * inv;
}

// ---------------------------------------------------------------------------
// K4: per (b,h): channel-softmax q fused with ctx = q . k^T
// block = 128 thr (4 waves), wave owns one 16x16 quadrant of the 32x32 ctx.
// q/k chunks staged fragment-ordered: [tile(dr2|ec2 x ks)][lane][e]
// ---------------------------------------------------------------------------
__global__ __launch_bounds__(128) void k_ctx(
    const float* __restrict__ kq_t, const float* __restrict__ k_t,
    float* __restrict__ ctx)
{
    __shared__ float raw[32][132];
    __shared__ __align__(32) unsigned short qf[8 * 512];  // [dr2*4+ks][512]
    __shared__ __align__(32) unsigned short kf[8 * 512];  // [ec2*4+ks][512]
    int tid = threadIdx.x, wv = tid >> 5, lane = tid & 31;
    int half = lane >> 4, l15 = lane & 15;
    int bh = blockIdx.x;
    const int dr = (wv >> 1) * 16, ec = (wv & 1) * 16;
    long base = (long)bh * DKH * NTOK;

    v8f acc = {0.f,0.f,0.f,0.f,0.f,0.f,0.f,0.f};
    for (int c0 = 0; c0 < NTOK; c0 += 128) {
        for (int i = tid; i < 32 * 128; i += 128) {
            int ch = i >> 7, tt = i & 127;
            long idx = base + (long)ch * NTOK + c0 + tt;
            raw[ch][tt] = kq_t[idx];
            kf[((ch >> 4) * 4 + (tt >> 5)) * 512 + bSwz(tt & 31, ch & 15)] = f2bfu(k_t[idx]);
        }
        __syncthreads();
        {
            int tt = tid;
            float mx = -1e30f;
            for (int ch = 0; ch < 32; ++ch) mx = fmaxf(mx, raw[ch][tt]);
            float sum = 0.f;
            for (int ch = 0; ch < 32; ++ch) sum += __expf(raw[ch][tt] - mx);
            float inv = 1.f / sum;
            for (int ch = 0; ch < 32; ++ch)
                qf[((ch >> 4) * 4 + (tt >> 5)) * 512 + aSwz(ch & 15, tt & 31)]
                    = f2bfu(__expf(raw[ch][tt] - mx) * inv);
        }
        __syncthreads();
        #pragma unroll
        for (int ks = 0; ks < 4; ++ks) {
            v16bf af = ldFrag(&qf[((dr >> 4) * 4 + ks) * 512], lane);
            v16bf bf = ldFrag(&kf[((ec >> 4) * 4 + ks) * 512], lane);
            acc = __builtin_amdgcn_wmma_f32_16x16x32_bf16(
                false, af, false, bf, (short)0, acc, false, false);
        }
        __syncthreads();
    }
    #pragma unroll
    for (int r = 0; r < 8; ++r) {
        int d = dr + r + half * 8, e = ec + l15;
        ctx[(bh * 32 + d) * 32 + e] = acc[r];
    }
}

// ---------------------------------------------------------------------------
// K5: Mt[b, f=h*32+e, o] = sum_d W[o,h*32+d]*ctx[b,h,d,e]
//     output PRE-SWIZZLED in WMMA B-order: Mts[b][kc(8)][ct(32)][512]
// block per (b,h), 256 thr (8 waves), output 32x512, single K=32 step
// ---------------------------------------------------------------------------
__global__ __launch_bounds__(256) void k_mt(
    const float* __restrict__ ctx, const float* __restrict__ wrep,
    unsigned short* __restrict__ Mts)
{
    __shared__ __align__(32) unsigned short aF[2 * 512];   // [rt2][512], A = ctx^T
    __shared__ __align__(32) unsigned short bF[32 * 512];  // [ct][512],  B = W_h^T
    int tid = threadIdx.x, wv = tid >> 5, lane = tid & 31;
    int half = lane >> 4, l15 = lane & 15;
    int bh = blockIdx.x, b = bh >> 3, h = bh & 7;

    for (int i = tid; i < 32 * 32; i += 256) {
        int e = i >> 5, d = i & 31;
        aF[(e >> 4) * 512 + aSwz(e & 15, d)] = f2bfu(ctx[(bh * 32 + d) * 32 + e]);
    }
    for (int i = tid; i < 32 * DD2; i += 256) {
        int d = i >> 9, o = i & (DD2 - 1);
        bF[(o >> 4) * 512 + bSwz(d, o & 15)] = f2bfu(wrep[o * DD + h * 32 + d]);
    }
    __syncthreads();

    int rt = (wv & 1) * 16;
    v16bf af = ldFrag(&aF[(wv & 1) * 512], lane);
    #pragma unroll
    for (int j = 0; j < 8; ++j) {
        int ct = (wv >> 1) * 8 + j;
        v16bf bf = ldFrag(&bF[ct * 512], lane);
        v8f acc = {0.f,0.f,0.f,0.f,0.f,0.f,0.f,0.f};
        acc = __builtin_amdgcn_wmma_f32_16x16x32_bf16(
            false, af, false, bf, (short)0, acc, false, false);
        #pragma unroll
        for (int r = 0; r < 8; ++r) {
            int kk = rt + r + half * 8;             // f = h*32 + kk  ->  kc == h
            long idx = (((long)b * 8 + h) * 32 + ct) * 512 + bSwz(kk, l15);
            Mts[idx] = f2bfu(acc[r]);
        }
    }
}

// ---------------------------------------------------------------------------
// K6: out = x1 + LN( n1 . Mt + b_rep )   [32768 x 512], K=256
// Tiles arrive pre-swizzled in global memory:
//   B tile (32KB/step): one TENSOR_LOAD_TO_LDS (TDM, TENSORcnt)
//   A tile (1KB/step):  GLOBAL_LOAD_ASYNC_TO_LDS_B128 (ASYNCcnt)
// block = 256 thr (8 waves): 16 rows x 512 cols
// ---------------------------------------------------------------------------
__global__ __launch_bounds__(256) void k_gemm2_ln_res(
    const unsigned short* __restrict__ n1s, const unsigned short* __restrict__ Mts,
    const float* __restrict__ rbias, const float* __restrict__ g,
    const float* __restrict__ bb, const float* __restrict__ x1,
    float* __restrict__ out)
{
    __shared__ __align__(32) unsigned short aF[512];
    __shared__ __align__(32) unsigned short bF[32 * 512];
    __shared__ float rowsum[16], rowsq[16], rmean[16], rrstd[16];

    int tid = threadIdx.x, wv = tid >> 5, lane = tid & 31;
    int half = lane >> 4, l15 = lane & 15;
    long m0 = (long)blockIdx.x * 16;
    int b = (int)(m0 >> 12);

    unsigned ldsA = (unsigned)(unsigned long long)(void*)aF;
    unsigned ldsB = (unsigned)(unsigned long long)(void*)bF;
    unsigned long long gA = (unsigned long long)(const void*)(n1s + (size_t)blockIdx.x * 8 * 512);
    unsigned long long gB = (unsigned long long)(const void*)(Mts + (size_t)b * 8 * 16384);

    v8f zero = {0.f,0.f,0.f,0.f,0.f,0.f,0.f,0.f};
    v8f acc[4];
    #pragma unroll
    for (int t = 0; t < 4; ++t) acc[t] = zero;

    for (int kc = 0; kc < 8; ++kc) {
        if (tid < 64)                                   // A tile: 1KB = 64 x 16B async
            async16(ldsA + tid * 16, gA, kc * 1024 + tid * 16);
        if (wv == 0)                                    // B tile: 32KB via TDM, wave 0 issues
            tdm_load_32k(ldsB, gB + (unsigned long long)kc * 32768ull);
        wait_async0();
        __syncthreads();
        v16bf af = ldFrag(aF, lane);
        #pragma unroll
        for (int t = 0; t < 4; ++t) {
            v16bf bf = ldFrag(&bF[(wv * 4 + t) * 512], lane);
            acc[t] = __builtin_amdgcn_wmma_f32_16x16x32_bf16(
                false, af, false, bf, (short)0, acc[t], false, false);
        }
        __syncthreads();
    }

    if (tid < 16) { rowsum[tid] = 0.f; rowsq[tid] = 0.f; }
    __syncthreads();

    float vals[4][8];
    #pragma unroll
    for (int t = 0; t < 4; ++t) {
        int o = wv * 64 + t * 16 + l15;
        #pragma unroll
        for (int r = 0; r < 8; ++r) vals[t][r] = acc[t][r] + rbias[o];
    }
    #pragma unroll
    for (int r = 0; r < 8; ++r) {
        int m = r + half * 8;
        float s = 0.f, s2 = 0.f;
        #pragma unroll
        for (int t = 0; t < 4; ++t) { float v = vals[t][r]; s += v; s2 += v * v; }
        atomicAdd(&rowsum[m], s);
        atomicAdd(&rowsq[m], s2);
    }
    __syncthreads();
    if (tid < 16) {
        float mu = rowsum[tid] / (float)DD2;
        float var = rowsq[tid] / (float)DD2 - mu * mu;
        rmean[tid] = mu; rrstd[tid] = rsqrtf(var + EPSV);
    }
    __syncthreads();
    #pragma unroll
    for (int t = 0; t < 4; ++t) {
        int o = wv * 64 + t * 16 + l15;
        #pragma unroll
        for (int r = 0; r < 8; ++r) {
            int m = r + half * 8;
            long gi = (m0 + m) * DD2 + o;
            float v = (vals[t][r] - rmean[m]) * rrstd[m] * g[o] + bb[o];
            out[gi] = x1[gi] + v;
        }
    }
}

// ---------------------------------------------------------------------------
extern "C" void kernel_launch(void* const* d_in, const int* in_sizes, int n_in,
                              void* d_out, int out_size, void* d_ws, size_t ws_size,
                              hipStream_t stream)
{
    const float* x1 = (const float*)d_in[0];
    const float* x2 = (const float*)d_in[1];
    const float* lw = (const float*)d_in[2];
    const float* lb = (const float*)d_in[3];
    const float* g1 = (const float*)d_in[4];
    const float* b1 = (const float*)d_in[5];
    const float* wr = (const float*)d_in[6];
    const float* br = (const float*)d_in[7];
    const float* ga = (const float*)d_in[8];
    const float* ba = (const float*)d_in[9];
    float* out = (float*)d_out;

    char* ws = (char*)d_ws;
    size_t off = 0;
    unsigned short* n1s = (unsigned short*)(ws + off); off += (size_t)BB * NTOK * DD * 2;   // 16 MB (swizzled)
    float* kq_t         = (float*)(ws + off);          off += (size_t)BB * DD * NTOK * 4;   // 33 MB
    float* k_t          = (float*)(ws + off);          off += (size_t)BB * DD * NTOK * 4;   // 33 MB
    float* ctx          = (float*)(ws + off);          off += (size_t)BB * NHEADS * DKH * DKH * 4;
    unsigned short* Mts = (unsigned short*)(ws + off); off += (size_t)BB * DD * DD2 * 2;    // 2 MB (swizzled)

    k_gemm1_ln    <<<(BB * NTOK) / 16, 128, 0, stream>>>(x1, lw, lb, g1, b1, n1s);
    k_ln2_t       <<<(BB * NTOK) / 8, 256, 0, stream>>>(x2, g1, b1, kq_t);
    k_softmax_tok <<<BB * DD,          256, 0, stream>>>(kq_t, k_t);
    k_ctx         <<<BB * NHEADS,      128, 0, stream>>>(kq_t, k_t, ctx);
    k_mt          <<<BB * NHEADS,      256, 0, stream>>>(ctx, wr, Mts);
    k_gemm2_ln_res<<<(BB * NTOK) / 16, 256, 0, stream>>>(n1s, Mts, br, ga, ba, x1, out);
}
